// Low_Rank_GRU_39290360824512
// MI455X (gfx1250) — matrix operand order repaired and checked by reading, not compiled, not run on hardware
//
#include <hip/hip_runtime.h>
#include <hip/hip_bf16.h>

#define B_ 128
#define T_ 512
#define I_ 64
#define H_ 512
#define R_ 4
#define O_ 64
#define NOISE_STD_ 0.05f
#define TAU_ 0.2f

typedef __attribute__((ext_vector_type(16))) __bf16 v16bf;
typedef __attribute__((ext_vector_type(8)))  float  v8f;
typedef __attribute__((ext_vector_type(4)))  float  f4;

// ---------------------------------------------------------------------------
// WMMA fragment loaders (fp32 memory -> bf16 fragments, wave32 layouts per
// CDNA5 ISA 7.12.2).
// A fragment (16x32, 16-bit): lane L holds row M=(L&15); regs 0-3 hold
// K = kb..kb+7, regs 4-7 hold K = kb+16..kb+23 where kb = (L>>4)*8.
// ---------------------------------------------------------------------------
__device__ __forceinline__ v16bf load_a_frag(const float* __restrict__ A,
                                             int lda, int m0, int k0, int lane) {
  const int row = m0 + (lane & 15);
  const int kb  = k0 + ((lane >> 4) << 3);
  const float* p = A + (size_t)row * lda + kb;
  f4 a0 = *(const f4*)(p + 0);
  f4 a1 = *(const f4*)(p + 4);
  f4 a2 = *(const f4*)(p + 16);
  f4 a3 = *(const f4*)(p + 20);
  v16bf r;
#pragma unroll
  for (int i = 0; i < 4; ++i) {
    r[i]      = (__bf16)a0[i];
    r[4 + i]  = (__bf16)a1[i];
    r[8 + i]  = (__bf16)a2[i];
    r[12 + i] = (__bf16)a3[i];
  }
  return r;
}

// B fragment (32x16, 16-bit): lane L holds column N=(L&15); K = kb..kb+7 and
// kb+16..kb+23 where kb = (L>>4)*8 (same order as A).  We want B = W^T where
// W is row-major [N,K], so the K values lie along W's row.
__device__ __forceinline__ v16bf load_bT_frag(const float* __restrict__ W,
                                              int ldw, int n0, int k0, int lane) {
  const int n  = n0 + (lane & 15);
  const int kb = k0 + ((lane >> 4) << 3);
  const float* p = W + (size_t)n * ldw + kb;
  f4 b0 = *(const f4*)(p + 0);
  f4 b1 = *(const f4*)(p + 4);
  f4 b2 = *(const f4*)(p + 16);
  f4 b3 = *(const f4*)(p + 20);
  v16bf r;
#pragma unroll
  for (int i = 0; i < 4; ++i) {
    r[i]      = (__bf16)b0[i];
    r[4 + i]  = (__bf16)b1[i];
    r[8 + i]  = (__bf16)b2[i];
    r[12 + i] = (__bf16)b3[i];
  }
  return r;
}

// ---------------------------------------------------------------------------
// Phase 1: input projections for one time-chunk.
// P[g][tl][b][h] = sum_i u[b, t0+tl, i] * W_g[h, i] + b_g[h]   (g = r,z,h)
// One wave computes one 16(M=b) x 16(N=h) tile, K=64 -> 2 WMMA.
// grid = (TC * B/16, 96/8), block = 256 (8 waves; each wave owns one n-tile).
// ---------------------------------------------------------------------------
__global__ __launch_bounds__(256) void proj_kernel(
    const float* __restrict__ u,
    const float* __restrict__ W_ir, const float* __restrict__ b_ir,
    const float* __restrict__ W_iz, const float* __restrict__ b_iz,
    const float* __restrict__ W_ih, const float* __restrict__ b_ih,
    float* __restrict__ P, int t0, int TC) {
  const int lane  = threadIdx.x & 31;
  const int wave  = threadIdx.x >> 5;
  const int mtile = blockIdx.x;            // 0 .. TC*8-1
  const int tl    = mtile >> 3;            // local timestep
  const int b0    = (mtile & 7) << 4;      // batch-row base
  const int nt    = blockIdx.y * 8 + wave; // 0..95
  const int g     = nt >> 5;               // gate 0/1/2
  const int h0    = (nt & 31) << 4;        // hidden base

  const float* W    = (g == 0) ? W_ir : (g == 1) ? W_iz : W_ih;
  const float* bias = (g == 0) ? b_ir : (g == 1) ? b_iz : b_ih;

  // A row index = batch b, lda = T*I, shifted to timestep t0+tl
  const float* A = u + (size_t)(t0 + tl) * I_;

  v8f acc = {};
#pragma unroll
  for (int k0 = 0; k0 < I_; k0 += 32) {
    v16bf a  = load_a_frag(A, T_ * I_, b0, k0, lane);
    v16bf bm = load_bT_frag(W, I_, h0, k0, lane);
    acc = __builtin_amdgcn_wmma_f32_16x16x32_bf16(false, a, false, bm,
                                                  (short)0, acc, false, false);
  }

  const int   col   = h0 + (lane & 15);
  const float bv    = bias[col];
  const int   rbase = b0 + ((lane >> 4) << 3);
  float* Pg = P + (size_t)g * TC * B_ * H_;
#pragma unroll
  for (int r = 0; r < 8; ++r) {
    const int m_local = tl * B_ + rbase + r;   // [tl][b] row
    Pg[(size_t)m_local * H_ + col] = acc[r] + bv;
  }
}

// ---------------------------------------------------------------------------
// Phase 2: sequential scan for one time-chunk.
// One workgroup per batch row; thread h owns x[b,h] in a register and the
// rank-4 N/M rows in registers.  Two block reductions per step, ONE barrier
// each: after the intra-wave butterfly, every wave redundantly reduces the
// 16 per-wave partials (LDS broadcast read + 4 shfl steps), so no separate
// broadcast barrier is needed and no wave0 serialization sits on the
// critical path.  Ping-pong LDS buffers make the single barrier race-free.
// ---------------------------------------------------------------------------
__device__ __forceinline__ float fast_sigmoid(float x) {
  return __builtin_amdgcn_rcpf(1.0f + __expf(-x));
}
__device__ __forceinline__ float fast_tanh(float x) {
  return 2.0f * __builtin_amdgcn_rcpf(1.0f + __expf(-2.0f * x)) - 1.0f;
}

template <int K>
__device__ __forceinline__ void block_reduce(float* v, float (&red)[16][9],
                                             int lane, int wave) {
  // intra-wave butterfly: every lane ends with the wave sum
#pragma unroll
  for (int off = 16; off >= 1; off >>= 1) {
#pragma unroll
    for (int k = 0; k < K; ++k) v[k] += __shfl_xor(v[k], off, 32);
  }
  if (lane == 0) {
#pragma unroll
    for (int k = 0; k < K; ++k) red[wave][k] = v[k];
  }
  __syncthreads();
  // every wave reduces the 16 wave-partials itself (redundant but parallel);
  // lanes 16..31 mirror 0..15 (same-address LDS broadcast, conflict-free)
#pragma unroll
  for (int k = 0; k < K; ++k) v[k] = red[lane & 15][k];
#pragma unroll
  for (int off = 8; off >= 1; off >>= 1) {
#pragma unroll
    for (int k = 0; k < K; ++k) v[k] += __shfl_xor(v[k], off, 32);
  }
  // all lanes now hold the full block sums in v[0..K-1]
}

__global__ __launch_bounds__(512) void scan_kernel(
    const float* __restrict__ P, const float* __restrict__ noise,
    const float* __restrict__ x0,
    const float* __restrict__ N_hr, const float* __restrict__ M_hr,
    const float* __restrict__ N_hz, const float* __restrict__ M_hz,
    const float* __restrict__ N_hh, const float* __restrict__ M_hh,
    float* __restrict__ traj, float* __restrict__ x_last,
    float* __restrict__ x_state, int t0, int TC) {
  const int b    = blockIdx.x;
  const int h    = threadIdx.x;
  const int lane = h & 31;
  const int wave = h >> 5;
  const float scale = 1.0f / (float)H_;

  const f4 nr = *(const f4*)(N_hr + 4 * h);
  const f4 mr = *(const f4*)(M_hr + 4 * h);
  const f4 nz = *(const f4*)(N_hz + 4 * h);
  const f4 mz = *(const f4*)(M_hz + 4 * h);
  const f4 nh = *(const f4*)(N_hh + 4 * h);
  const f4 mh = *(const f4*)(M_hh + 4 * h);

  float x = (t0 == 0) ? x0[(size_t)b * H_ + h] : x_state[(size_t)b * H_ + h];

  const float* Pr = P;
  const float* Pz = P + (size_t)TC * B_ * H_;
  const float* Ph = P + 2 * (size_t)TC * B_ * H_;

  __shared__ float redA[16][9];
  __shared__ float redB[16][9];

  const size_t step = (size_t)B_ * H_;                 // per-timestep stride
  size_t pi = (size_t)b * H_ + h;                      // local index, tl = 0
  size_t ni = ((size_t)t0 * B_ + b) * H_ + h;          // noise index, tl = 0

  // software pipeline: current-step operands held in registers
  float ir  = Pr[pi], iz = Pz[pi], ih = Ph[pi];
  float nse = noise[ni];

  for (int tl = 0; tl < TC; ++tl) {
    // ---- issue next step's loads early (hidden behind the barriers) ----
    const size_t dn  = (tl + 1 < TC) ? step : 0;
    const float ir_n = Pr[pi + dn];
    const float iz_n = Pz[pi + dn];
    const float ih_n = Ph[pi + dn];
    const float ns_n = noise[ni + dn];
    if (tl + 4 < TC) {
      __builtin_prefetch(Pr + pi + 4 * step, 0, 2);
      __builtin_prefetch(Pz + pi + 4 * step, 0, 2);
      __builtin_prefetch(Ph + pi + 4 * step, 0, 2);
      __builtin_prefetch(noise + ni + 4 * step, 0, 2);
    }

    // ---- gate pre-activations from the low-rank recurrent term ----
    float v[8];
#pragma unroll
    for (int r = 0; r < 4; ++r) { v[r] = x * nr[r]; v[4 + r] = x * nz[r]; }
    block_reduce<8>(v, redA, lane, wave);

    const float hr = scale * (v[0] * mr[0] + v[1] * mr[1] +
                              v[2] * mr[2] + v[3] * mr[3]);
    const float hz = scale * (v[4] * mz[0] + v[5] * mz[1] +
                              v[6] * mz[2] + v[7] * mz[3]);
    const float rg = fast_sigmoid(ir + hr);
    const float zg = fast_sigmoid(iz + hz);

    const float rx = rg * x;
    float w[4];
#pragma unroll
    for (int r = 0; r < 4; ++r) w[r] = rx * nh[r];
    block_reduce<4>(w, redB, lane, wave);

    const float hh = scale * (w[0] * mh[0] + w[1] * mh[1] +
                              w[2] * mh[2] + w[3] * mh[3]);
    const float g   = fast_tanh(ih + hh);
    const float rec = (1.0f - zg) * g + zg * x;
    x = x + NOISE_STD_ * nse + TAU_ * (rec - x);

    traj[((size_t)b * T_ + t0 + tl) * H_ + h] = x;

    ir = ir_n; iz = iz_n; ih = ih_n; nse = ns_n;
    pi += dn; ni += dn;
  }

  x_state[(size_t)b * H_ + h] = x;
  x_last[(size_t)b * H_ + h] = x;  // final chunk leaves the correct value
}

// ---------------------------------------------------------------------------
// Phase 3: output projection.  out[m, o] = sum_h traj[m, h] * W_out[o, h] + b_out[o]
// with m = b*T + t (traj rows are contiguous, lda = H).  One wave per 16x16
// tile, K = 512 -> 16 WMMA.  Block = 8 waves = 2 m-tiles x 4 n-tiles.
// ---------------------------------------------------------------------------
__global__ __launch_bounds__(256) void out_kernel(
    const float* __restrict__ traj, const float* __restrict__ W_out,
    const float* __restrict__ b_out, float* __restrict__ out) {
  const int lane  = threadIdx.x & 31;
  const int wave  = threadIdx.x >> 5;
  const int mtile = blockIdx.x * 2 + (wave >> 2);
  const int ntile = wave & 3;
  const int m0 = mtile << 4;
  const int n0 = ntile << 4;

  v8f acc = {};
#pragma unroll 8
  for (int k0 = 0; k0 < H_; k0 += 32) {
    v16bf a  = load_a_frag(traj, H_, m0, k0, lane);
    v16bf bm = load_bT_frag(W_out, H_, n0, k0, lane);
    acc = __builtin_amdgcn_wmma_f32_16x16x32_bf16(false, a, false, bm,
                                                  (short)0, acc, false, false);
  }

  const int   col   = n0 + (lane & 15);
  const float bv    = b_out[col];
  const int   rbase = m0 + ((lane >> 4) << 3);
#pragma unroll
  for (int r = 0; r < 8; ++r)
    out[(size_t)(rbase + r) * O_ + col] = acc[r] + bv;
}

// ---------------------------------------------------------------------------
extern "C" void kernel_launch(void* const* d_in, const int* in_sizes, int n_in,
                              void* d_out, int out_size, void* d_ws, size_t ws_size,
                              hipStream_t stream) {
  (void)in_sizes; (void)n_in; (void)out_size;

  const float* u     = (const float*)d_in[0];
  const float* x0    = (const float*)d_in[1];
  const float* noise = (const float*)d_in[2];
  const float* W_ir  = (const float*)d_in[3];
  const float* b_ir  = (const float*)d_in[4];
  const float* W_iz  = (const float*)d_in[5];
  const float* b_iz  = (const float*)d_in[6];
  const float* W_ih  = (const float*)d_in[7];
  const float* b_ih  = (const float*)d_in[8];
  const float* N_hr  = (const float*)d_in[9];
  const float* M_hr  = (const float*)d_in[10];
  const float* N_hz  = (const float*)d_in[11];
  const float* M_hz  = (const float*)d_in[12];
  const float* N_hh  = (const float*)d_in[13];
  const float* M_hh  = (const float*)d_in[14];
  const float* W_out = (const float*)d_in[15];
  const float* b_out = (const float*)d_in[16];

  // Outputs, concatenated flat in return order: output, x_last, traj.
  float* out    = (float*)d_out;
  float* x_last = out + (size_t)B_ * T_ * O_;
  float* traj   = x_last + (size_t)B_ * H_;

  // Workspace: [3, TC, B, H] projections + [B, H] persistent hidden state.
  // TC picked from ws_size (fixed per process -> deterministic schedule).
  const size_t state_f = (size_t)B_ * H_;
  int TC = 64;
  while (TC > 1 &&
         ((size_t)3 * TC * B_ * H_ + state_f) * sizeof(float) > ws_size)
    TC >>= 1;

  float* P       = (float*)d_ws;
  float* x_state = P + (size_t)3 * TC * B_ * H_;

  for (int t0 = 0; t0 < T_; t0 += TC) {
    dim3 pgrid(TC * (B_ / 16), (3 * H_ / 16) / 8);
    proj_kernel<<<pgrid, 256, 0, stream>>>(u, W_ir, b_ir, W_iz, b_iz,
                                           W_ih, b_ih, P, t0, TC);
    scan_kernel<<<B_, H_, 0, stream>>>(P, noise, x0, N_hr, M_hr, N_hz, M_hz,
                                       N_hh, M_hh, traj, x_last, x_state,
                                       t0, TC);
  }

  out_kernel<<<(B_ * T_ / 16) / 2, 256, 0, stream>>>(traj, W_out, b_out, out);
}
